// Model_40596030882213
// MI455X (gfx1250) — compile-verified, matrix operand order checked by reading, not run on hardware
//
#include <hip/hip_runtime.h>
#include <hip/hip_bf16.h>

// ---------------- model constants ----------------
#define Bh      32
#define SEQL    512
#define LBL     256
#define PRD     256
#define DECL    512          // DEC_LEN
#define CINc    21
#define DD      512
#define DFFc    2048
#define TFEATc  4
#define TOPK    18           // int(3 * ln(512))
#define MTOK    (Bh * SEQL)  // 16384 tokens (enc and dec have same length)

typedef unsigned short ushort_t;
typedef __attribute__((ext_vector_type(16))) __bf16 v16bf;
typedef __attribute__((ext_vector_type(8)))  float  v8f;
typedef __attribute__((ext_vector_type(4)))  unsigned int u32x4;
typedef __attribute__((ext_vector_type(4)))  int i32x4;

union FragBF { u32x4 q[2]; v16bf v; };

// ---- CDNA5 async global->LDS path (guarded; falls back to sync copy) ----
#if defined(__AMDGCN__) && __has_builtin(__builtin_amdgcn_global_load_async_to_lds_b128)
#define HAS_ASYNC_LDS 1
#else
#define HAS_ASYNC_LDS 0
#endif

typedef __attribute__((address_space(1))) i32x4* g_i32x4_ptr;
typedef __attribute__((address_space(3))) i32x4* l_i32x4_ptr;

__device__ __forceinline__ void copy16_g2l(const ushort_t* g, ushort_t* l) {
#if HAS_ASYNC_LDS
    __builtin_amdgcn_global_load_async_to_lds_b128(
        (g_i32x4_ptr)(unsigned long long)g,
        (l_i32x4_ptr)(unsigned int)(unsigned long long)l,
        0, 0);
#else
    *(u32x4*)l = *(const u32x4*)g;
#endif
}

template <int N> __device__ __forceinline__ void wait_async() {
#if HAS_ASYNC_LDS
#if __has_builtin(__builtin_amdgcn_s_wait_asynccnt)
    __builtin_amdgcn_s_wait_asynccnt(N);
#else
    asm volatile("s_wait_asynccnt %0" :: "n"(N) : "memory");
#endif
#endif
}

// ---------------- f32 -> bf16 (RNE) ----------------
__device__ __forceinline__ ushort_t f2bf(float f) {
    unsigned int u = __float_as_uint(f);
    return (ushort_t)((u + 0x7FFFu + ((u >> 16) & 1u)) >> 16);
}

__global__ void f32_to_bf16_kernel(const float* __restrict__ src,
                                   ushort_t* __restrict__ dst, int n) {
    int i = blockIdx.x * blockDim.x + threadIdx.x;
    if (i < n) dst[i] = f2bf(src[i]);
}

// convert weight [K][N] f32 -> transposed bf16 [N][K] (so B tiles are
// contiguous 16B chunks -> async-LDS eligible, no LDS scatter)
__global__ void f32_to_bf16_tr_kernel(const float* __restrict__ src,
                                      ushort_t* __restrict__ dst, int K, int N) {
    int i = blockIdx.x * blockDim.x + threadIdx.x;
    if (i >= K * N) return;
    int k = i / N, n = i % N;               // coalesced read
    dst[(size_t)n * K + k] = f2bf(src[i]);  // strided write (weights are tiny)
}

// ---------------- WMMA GEMM: C[M,N] = A[M,K] @ Wt[N,K]^T (+bias)(+gelu) ---
// A row-major bf16 [M][K]; Wt TRANSPOSED bf16 [N][K]. M%64==0,N%128==0,K%32==0.
// 256 threads = 8 wave32; block tile 64x128; wave tile 32x32 (2x2 WMMA tiles).
// Double-buffered LDS staged via GLOBAL_LOAD_ASYNC_TO_LDS_B128 (ASYNCcnt).
#define LDT 40   // halves per LDS row: 80B row stride, 16B aligned chunks,
                 // conflict-free b128 reads (20-dword stride spans 64 banks)
__global__ __launch_bounds__(256) void wmma_gemm_bf16(
    const ushort_t* __restrict__ A, const ushort_t* __restrict__ Wt,
    const float* __restrict__ bias, float* __restrict__ Cf,
    ushort_t* __restrict__ Cbf, int M, int N, int K, int fuse_gelu)
{
    __shared__ __align__(16) ushort_t As[2][64 * LDT];
    __shared__ __align__(16) ushort_t Bs[2][128 * LDT];

    const int tid  = threadIdx.x;
    const int wave = tid >> 5, lane = tid & 31;
    const int m0 = blockIdx.y * 64;
    const int n0 = blockIdx.x * 128;
    const int wm = (wave >> 2) * 32;   // 0 / 32
    const int wn = (wave & 3) * 32;    // 0 / 32 / 64 / 96
    const int lm = lane & 15, half = lane >> 4;

    // per-thread staging coordinates (16B chunks of 8 halves)
    const int rA = (tid * 8) >> 5;          // 0..63
    const int cA = (tid * 8) & 31;          // 0,8,16,24
    const int rB0 = (tid * 2)     >> 2;     // 0..127
    const int cB0 = ((tid * 2)     & 3) * 8;
    const int rB1 = (tid * 2 + 1) >> 2;
    const int cB1 = ((tid * 2 + 1) & 3) * 8;

    v8f acc[2][2] = {};

    auto stage = [&](int k0, int buf) {
        copy16_g2l(A  + (size_t)(m0 + rA)  * K + (k0 + cA),  &As[buf][rA  * LDT + cA]);
        copy16_g2l(Wt + (size_t)(n0 + rB0) * K + (k0 + cB0), &Bs[buf][rB0 * LDT + cB0]);
        copy16_g2l(Wt + (size_t)(n0 + rB1) * K + (k0 + cB1), &Bs[buf][rB1 * LDT + cB1]);
    };

    const int nkt = K >> 5;
    stage(0, 0);                       // prologue: 3 async ops outstanding
    for (int kt = 0; kt < nkt; ++kt) {
        const int cur = kt & 1;
        if (kt + 1 < nkt) {
            stage((kt + 1) << 5, cur ^ 1);  // overlap next tile's DMA
            wait_async<3>();                // current tile done (in-order)
        } else {
            wait_async<0>();
        }
        __syncthreads();               // all waves' tiles visible

        // fragments per documented 16-bit A layout:
        // lane m = lane%16, half = lane/16 -> K halves {8h..8h+7},{16+8h..}
        FragBF af[2], bf[2];
#pragma unroll
        for (int mt = 0; mt < 2; ++mt) {
            const ushort_t* row = &As[cur][(wm + 16 * mt + lm) * LDT];
            af[mt].q[0] = *(const u32x4*)(row + 8 * half);
            af[mt].q[1] = *(const u32x4*)(row + 16 + 8 * half);
        }
#pragma unroll
        for (int nt = 0; nt < 2; ++nt) {
            const ushort_t* row = &Bs[cur][(wn + 16 * nt + lm) * LDT];
            bf[nt].q[0] = *(const u32x4*)(row + 8 * half);
            bf[nt].q[1] = *(const u32x4*)(row + 16 + 8 * half);
        }
#pragma unroll
        for (int mt = 0; mt < 2; ++mt)
#pragma unroll
            for (int nt = 0; nt < 2; ++nt)
                acc[mt][nt] = __builtin_amdgcn_wmma_f32_16x16x32_bf16(
                    false, af[mt].v, false, bf[nt].v,
                    (short)0, acc[mt][nt], false, false);
        __syncthreads();               // done reading before buffer reuse
    }

    // epilogue: C layout lane -> N = lane%16, VGPR j -> M = j + 8*(lane/16)
#pragma unroll
    for (int mt = 0; mt < 2; ++mt) {
#pragma unroll
        for (int nt = 0; nt < 2; ++nt) {
            int n = n0 + wn + 16 * nt + lm;
            float bv = bias ? bias[n] : 0.0f;
#pragma unroll
            for (int j = 0; j < 8; ++j) {
                int m = m0 + wm + 16 * mt + j + 8 * half;
                float v = acc[mt][nt][j] + bv;
                if (fuse_gelu) v = 0.5f * v * (1.0f + erff(v * 0.7071067811865475f));
                if (Cbf) Cbf[(size_t)m * N + n] = f2bf(v);  // fused downconvert
                else     Cf [(size_t)m * N + n] = v;
            }
        }
    }
}

// ---------------- embedding: circular conv3 over Cin + time matmul --------
__global__ __launch_bounds__(256) void embed_kernel(
    const float* __restrict__ x, const float* __restrict__ xm,
    const float* __restrict__ convw, const float* __restrict__ timew,
    float* __restrict__ out, int L, int Cin)
{
    int bl = blockIdx.x, b = bl / L, l = bl % L;
    int lm1 = (l == 0) ? L - 1 : l - 1;
    int lp1 = (l == L - 1) ? 0 : l + 1;
    __shared__ float xs[3][32];
    __shared__ float ms[TFEATc];
    if (threadIdx.x < Cin) {
        xs[0][threadIdx.x] = x[((size_t)b * L + lm1) * Cin + threadIdx.x];
        xs[1][threadIdx.x] = x[((size_t)b * L + l  ) * Cin + threadIdx.x];
        xs[2][threadIdx.x] = x[((size_t)b * L + lp1) * Cin + threadIdx.x];
    }
    if (threadIdx.x < TFEATc)
        ms[threadIdx.x] = xm[((size_t)b * L + l) * TFEATc + threadIdx.x];
    __syncthreads();
    for (int d = threadIdx.x; d < DD; d += blockDim.x) {
        const float* w = convw + (size_t)d * Cin * 3;
        float acc = 0.0f;
        for (int c = 0; c < Cin; ++c)
            acc += xs[0][c] * w[c*3+0] + xs[1][c] * w[c*3+1] + xs[2][c] * w[c*3+2];
        for (int t = 0; t < TFEATc; ++t) acc += ms[t] * timew[t * DD + d];
        out[((size_t)b * L + l) * DD + d] = acc;
    }
}

// ---------------- moving-average (k=25, edge-replicate) decomposition -----
__global__ void movavg_decomp_kernel(const float* __restrict__ x,
                                     float* __restrict__ seas,
                                     float* __restrict__ trend, int L, int C)
{
    int i = blockIdx.x * blockDim.x + threadIdx.x;
    int total = Bh * L * C;
    if (i >= total) return;
    int c = i % C, l = (i / C) % L, b = i / (C * L);
    const float* xb = x + (size_t)b * L * C + c;
    float s = 0.0f;
    for (int j = -12; j <= 12; ++j) {
        int t = l + j;
        t = t < 0 ? 0 : (t >= L ? L - 1 : t);
        s += xb[(size_t)t * C];
    }
    float m = s * (1.0f / 25.0f);
    trend[i] = m;
    seas[i]  = x[i] - m;
}

// ---------------- per-(b,c) mean over time ----------------
__global__ void mean_time_kernel(const float* __restrict__ x,
                                 float* __restrict__ mean, int L, int C)
{
    int i = blockIdx.x * blockDim.x + threadIdx.x;
    if (i >= Bh * C) return;
    int b = i / C, c = i % C;
    float s = 0.0f;
    for (int l = 0; l < L; ++l) s += x[((size_t)b * L + l) * C + c];
    mean[i] = s / (float)L;
}

// ---------------- build trend_init / seasonal_init ----------------
__global__ void build_init_kernel(const float* __restrict__ seasF,
                                  const float* __restrict__ trendF,
                                  const float* __restrict__ mean,
                                  float* __restrict__ seas_init,
                                  float* __restrict__ trend_init)
{
    int i = blockIdx.x * blockDim.x + threadIdx.x;
    int total = Bh * DECL * CINc;
    if (i >= total) return;
    int c = i % CINc, l = (i / CINc) % DECL, b = i / (CINc * DECL);
    if (l < LBL) {
        int src = ((size_t)b * SEQL + (SEQL - LBL + l)) * CINc + c;
        seas_init[i]  = seasF[src];
        trend_init[i] = trendF[src];
    } else {
        seas_init[i]  = 0.0f;
        trend_init[i] = mean[b * CINc + c];
    }
}

// ---------------- mean circular correlation over all channels -------------
// corr[b,tau] = (1/D) * sum_{t,d} q[b,(t+tau)%L,d] * k[b,t,d]
__global__ __launch_bounds__(256) void mean_corr_kernel(
    const float* __restrict__ q, const float* __restrict__ k,
    float* __restrict__ corr, int L)
{
    int b = blockIdx.y, tau = blockIdx.x;
    const float* qb = q + (size_t)b * L * DD;
    const float* kb = k + (size_t)b * L * DD;
    float acc = 0.0f;
    for (int t = threadIdx.x; t < L; t += 256) {
        int ts = t + tau; if (ts >= L) ts -= L;
        const float4* qr = (const float4*)(qb + (size_t)ts * DD);
        const float4* kr = (const float4*)(kb + (size_t)t  * DD);
        for (int d4 = 0; d4 < DD / 4; ++d4) {
            float4 a = qr[d4], c2 = kr[d4];
            acc += a.x * c2.x + a.y * c2.y + a.z * c2.z + a.w * c2.w;
        }
    }
    __shared__ float red[256];
    red[threadIdx.x] = acc; __syncthreads();
    for (int s = 128; s > 0; s >>= 1) {
        if (threadIdx.x < s) red[threadIdx.x] += red[threadIdx.x + s];
        __syncthreads();
    }
    if (threadIdx.x == 0) corr[(size_t)b * L + tau] = red[0] / (float)DD;
}

// ---------------- per-batch top-k + softmax ----------------
__global__ __launch_bounds__(256) void topk_softmax_kernel(
    const float* __restrict__ corr, int* __restrict__ delays,
    float* __restrict__ weights, int L)
{
    int b = blockIdx.x;
    __shared__ float vals[SEQL];
    __shared__ float rmax[256];
    __shared__ int   ridx[256];
    __shared__ float wsel[TOPK];
    __shared__ int   dsel[TOPK];
    for (int i = threadIdx.x; i < L; i += 256) vals[i] = corr[(size_t)b * L + i];
    __syncthreads();
    for (int it = 0; it < TOPK; ++it) {
        float best = -1e30f; int bi = 0;
        for (int i = threadIdx.x; i < L; i += 256)
            if (vals[i] > best) { best = vals[i]; bi = i; }
        rmax[threadIdx.x] = best; ridx[threadIdx.x] = bi; __syncthreads();
        for (int s = 128; s > 0; s >>= 1) {
            if (threadIdx.x < s && rmax[threadIdx.x + s] > rmax[threadIdx.x]) {
                rmax[threadIdx.x] = rmax[threadIdx.x + s];
                ridx[threadIdx.x] = ridx[threadIdx.x + s];
            }
            __syncthreads();
        }
        if (threadIdx.x == 0) { wsel[it] = rmax[0]; dsel[it] = ridx[0]; vals[ridx[0]] = -1e30f; }
        __syncthreads();
    }
    if (threadIdx.x == 0) {
        float mx = wsel[0], ssum = 0.0f;
        for (int i = 0; i < TOPK; ++i) { float e = __expf(wsel[i] - mx); wsel[i] = e; ssum += e; }
        for (int i = 0; i < TOPK; ++i) {
            weights[b * TOPK + i] = wsel[i] / ssum;
            delays [b * TOPK + i] = dsel[i];
        }
    }
}

// ---------------- weighted circular gather of V ----------------
__global__ __launch_bounds__(256) void agg_kernel(
    const float* __restrict__ v, const int* __restrict__ delays,
    const float* __restrict__ w, float* __restrict__ out, int L)
{
    int bl = blockIdx.x, b = bl / L, l = bl % L;
    __shared__ int dsh[TOPK]; __shared__ float wsh[TOPK];
    if (threadIdx.x < TOPK) {
        dsh[threadIdx.x] = delays[b * TOPK + threadIdx.x];
        wsh[threadIdx.x] = w[b * TOPK + threadIdx.x];
    }
    __syncthreads();
    const float* vb = v + (size_t)b * L * DD;
    for (int d = threadIdx.x; d < DD; d += 256) {
        float acc = 0.0f;
        for (int i = 0; i < TOPK; ++i) {
            int ls = l + dsh[i]; if (ls >= L) ls -= L;
            acc += wsh[i] * vb[(size_t)ls * DD + d];
        }
        out[((size_t)b * L + l) * DD + d] = acc;
    }
}

// ---------------- elementwise helpers ----------------
__global__ void add_kernel(const float* a, const float* b, float* o, int n) {
    int i = blockIdx.x * blockDim.x + threadIdx.x;
    if (i < n) o[i] = a[i] + b[i];
}
__global__ void add3_kernel(const float* a, const float* b, const float* c, float* o, int n) {
    int i = blockIdx.x * blockDim.x + threadIdx.x;
    if (i < n) o[i] = a[i] + b[i] + c[i];
}

// ---------------- layernorm over D ----------------
__global__ __launch_bounds__(256) void layernorm_kernel(
    const float* __restrict__ x, const float* __restrict__ g,
    const float* __restrict__ bb, float* __restrict__ out)
{
    int row = blockIdx.x;
    const float* xr = x + (size_t)row * DD;
    float s = 0.0f, s2 = 0.0f;
    for (int d = threadIdx.x; d < DD; d += 256) { float v = xr[d]; s += v; s2 += v * v; }
    __shared__ float r1[256], r2[256];
    r1[threadIdx.x] = s; r2[threadIdx.x] = s2; __syncthreads();
    for (int st = 128; st > 0; st >>= 1) {
        if (threadIdx.x < st) { r1[threadIdx.x] += r1[threadIdx.x+st]; r2[threadIdx.x] += r2[threadIdx.x+st]; }
        __syncthreads();
    }
    float mu  = r1[0] / (float)DD;
    float var = r2[0] / (float)DD - mu * mu;
    float inv = rsqrtf(var + 1e-5f);
    for (int d = threadIdx.x; d < DD; d += 256)
        out[(size_t)row * DD + d] = (xr[d] - mu) * inv * g[d] + bb[d];
}

// ---------------- subtract per-(b,d) mean over sequence ----------------
__global__ void sub_colmean_kernel(const float* __restrict__ x,
                                   float* __restrict__ out, int L)
{
    int i = blockIdx.x * blockDim.x + threadIdx.x;
    if (i >= Bh * DD) return;
    int b = i / DD, d = i % DD;
    const float* xb = x + (size_t)b * L * DD + d;
    float s = 0.0f;
    for (int l = 0; l < L; ++l) s += xb[(size_t)l * DD];
    s /= (float)L;
    float* ob = out + (size_t)b * L * DD + d;
    for (int l = 0; l < L; ++l) ob[(size_t)l * DD] = xb[(size_t)l * DD] - s;
}

// ---------------- trend conv: D -> CIN, circular conv3 ----------------
__global__ void trend_conv_kernel(const float* __restrict__ rt,
                                  const float* __restrict__ w,
                                  float* __restrict__ out, int L)
{
    int bl = blockIdx.x, b = bl / L, l = bl % L;
    int lm1 = (l == 0) ? L - 1 : l - 1;
    int lp1 = (l == L - 1) ? 0 : l + 1;
    const float* r0 = rt + ((size_t)b * L + lm1) * DD;
    const float* r1 = rt + ((size_t)b * L + l  ) * DD;
    const float* r2 = rt + ((size_t)b * L + lp1) * DD;
    int c = threadIdx.x;
    if (c < CINc) {
        const float* wc = w + (size_t)c * DD * 3;
        float acc = 0.0f;
        for (int d = 0; d < DD; ++d)
            acc += r0[d] * wc[d*3+0] + r1[d] * wc[d*3+1] + r2[d] * wc[d*3+2];
        out[((size_t)b * L + l) * CINc + c] = acc;
    }
}

// ---------------- final: trend_out + dec @ proj_w + proj_b, last PRED -----
__global__ void final_kernel(const float* __restrict__ dec,
                             const float* __restrict__ trend_out,
                             const float* __restrict__ pw,
                             const float* __restrict__ pb,
                             float* __restrict__ out)
{
    int bp = blockIdx.x, b = bp / PRD, p = bp % PRD;
    int l = LBL + p;
    int c = threadIdx.x;
    if (c < CINc) {
        const float* dr = dec + ((size_t)b * DECL + l) * DD;
        float acc = pb[c];
        for (int d = 0; d < DD; ++d) acc += dr[d] * pw[d * CINc + c];
        acc += trend_out[((size_t)b * DECL + l) * CINc + c];
        out[((size_t)b * PRD + p) * CINc + c] = acc;
    }
}

// ========================= host orchestration =========================
extern "C" void kernel_launch(void* const* d_in, const int* in_sizes, int n_in,
                              void* d_out, int out_size, void* d_ws, size_t ws_size,
                              hipStream_t stream) {
    (void)in_sizes; (void)n_in; (void)out_size; (void)ws_size;

    const float* x_enc       = (const float*)d_in[0];
    const float* x_mark_enc  = (const float*)d_in[1];
    const float* x_mark_dec  = (const float*)d_in[3];
    const float* enc_conv_w  = (const float*)d_in[4];
    const float* enc_time_w  = (const float*)d_in[5];
    const float* dec_conv_w  = (const float*)d_in[6];
    const float* dec_time_w  = (const float*)d_in[7];
    const float* enc_attn_w  = (const float*)d_in[8];
    const float* enc_attn_b  = (const float*)d_in[9];
    const float* enc_ff1_w   = (const float*)d_in[10];
    const float* enc_ff2_w   = (const float*)d_in[11];
    const float* enc_norm_g  = (const float*)d_in[12];
    const float* enc_norm_b  = (const float*)d_in[13];
    const float* dec_self_w  = (const float*)d_in[14];
    const float* dec_self_b  = (const float*)d_in[15];
    const float* dec_cross_w = (const float*)d_in[16];
    const float* dec_cross_b = (const float*)d_in[17];
    const float* dec_ff1_w   = (const float*)d_in[18];
    const float* dec_ff2_w   = (const float*)d_in[19];
    const float* dec_trend_w = (const float*)d_in[20];
    const float* dec_norm_g  = (const float*)d_in[21];
    const float* dec_norm_b  = (const float*)d_in[22];
    const float* proj_w      = (const float*)d_in[23];
    const float* proj_b      = (const float*)d_in[24];
    float* out = (float*)d_out;

    // ---- bump allocator over d_ws ----
    char* wsb = (char*)d_ws;
    size_t off = 0;
    auto alloc = [&](size_t bytes) -> void* {
        void* p = wsb + off;
        off += (bytes + 255) & ~(size_t)255;
        return p;
    };

    const size_t S = (size_t)MTOK * DD;            // activation elements
    ushort_t* bfA = (ushort_t*)alloc(S * 2);                   // staged A (bf16)
    ushort_t* bfW = (ushort_t*)alloc((size_t)DD * DFFc * 2);   // staged Wt (bf16)
    ushort_t* bfH = (ushort_t*)alloc((size_t)MTOK * DFFc * 2); // FFN hidden (bf16)
    float* X   = (float*)alloc(S * 4);
    float* Q   = (float*)alloc(S * 4);
    float* Kb  = (float*)alloc(S * 4);
    float* Vb  = (float*)alloc(S * 4);
    float* TMP = (float*)alloc(S * 4);
    float* TRJ = (float*)alloc(S * 4);
    float* ENC = (float*)alloc(S * 4);
    float* T1  = (float*)alloc(S * 4);
    float* T2  = (float*)alloc(S * 4);
    float* T3  = (float*)alloc(S * 4);
    float* corr    = (float*)alloc((size_t)Bh * SEQL * 4);
    int*   delays  = (int*)  alloc((size_t)Bh * TOPK * 4);
    float* weights = (float*)alloc((size_t)Bh * TOPK * 4);
    float* meanE   = (float*)alloc((size_t)Bh * CINc * 4);
    float* seasF   = (float*)alloc((size_t)Bh * SEQL * CINc * 4);
    float* trendF  = (float*)alloc((size_t)Bh * SEQL * CINc * 4);
    float* seasI   = (float*)alloc((size_t)Bh * DECL * CINc * 4);
    float* trendI  = (float*)alloc((size_t)Bh * DECL * CINc * 4);
    float* RT      = (float*)alloc((size_t)Bh * DECL * CINc * 4);

    auto cvt = [&](const float* src, ushort_t* dst, size_t n) {
        f32_to_bf16_kernel<<<(int)((n + 255) / 256), 256, 0, stream>>>(src, dst, (int)n);
    };
    auto cvtWtr = [&](const float* w, int K, int N) {
        f32_to_bf16_tr_kernel<<<(K * N + 255) / 256, 256, 0, stream>>>(w, bfW, K, N);
    };
    auto run = [&](const ushort_t* Abf, const float* bias, float* Cf, ushort_t* Cbf,
                   int M, int N, int K, int gelu) {
        dim3 grid(N / 128, M / 64);
        wmma_gemm_bf16<<<grid, 256, 0, stream>>>(Abf, bfW, bias, Cf, Cbf, M, N, K, gelu);
    };
    auto decompL = [&](const float* src, float* seas, float* trend, int L, int C) {
        int total = Bh * L * C;
        movavg_decomp_kernel<<<(total + 255) / 256, 256, 0, stream>>>(src, seas, trend, L, C);
    };
    auto addv = [&](const float* a, const float* b, float* o, size_t n) {
        add_kernel<<<(int)((n + 255) / 256), 256, 0, stream>>>(a, b, o, (int)n);
    };
    // autocorrelation attention block: out = Proj(agg(Q(xq),K(xkv),V(xkv)))
    auto autocorr = [&](const float* xq, const float* xkv,
                        const float* w4, const float* b4, float* outb) {
        cvt(xq, bfA, S);                                  // stage A once for Q
        cvtWtr(w4 + 0 * DD * DD, DD, DD); run(bfA, b4 + 0 * DD, Q, nullptr, MTOK, DD, DD, 0);
        if (xkv != xq) cvt(xkv, bfA, S);                  // restage A for K/V
        cvtWtr(w4 + 1 * DD * DD, DD, DD); run(bfA, b4 + 1 * DD, Kb, nullptr, MTOK, DD, DD, 0);
        cvtWtr(w4 + 2 * DD * DD, DD, DD); run(bfA, b4 + 2 * DD, Vb, nullptr, MTOK, DD, DD, 0);
        mean_corr_kernel<<<dim3(SEQL, Bh), 256, 0, stream>>>(Q, Kb, corr, SEQL);
        topk_softmax_kernel<<<Bh, 256, 0, stream>>>(corr, delays, weights, SEQL);
        agg_kernel<<<Bh * SEQL, 256, 0, stream>>>(Vb, delays, weights, TMP, SEQL);
        cvt(TMP, bfA, S);
        cvtWtr(w4 + 3 * DD * DD, DD, DD); run(bfA, b4 + 3 * DD, outb, nullptr, MTOK, DD, DD, 0);
    };
    // FFN: hidden kept in bf16 (fused downconvert), no f32 round-trip
    auto ffn = [&](const float* xin, const float* f1, const float* f2, float* outb) {
        cvt(xin, bfA, S);
        cvtWtr(f1, DD, DFFc);  run(bfA, nullptr, nullptr, bfH, MTOK, DFFc, DD, 1);
        cvtWtr(f2, DFFc, DD);  run(bfH, nullptr, outb, nullptr, MTOK, DD, DFFc, 0);
    };

    // ---- phase 0: series decomposition / init ----
    decompL(x_enc, seasF, trendF, SEQL, CINc);
    mean_time_kernel<<<(Bh * CINc + 255) / 256, 256, 0, stream>>>(x_enc, meanE, SEQL, CINc);
    build_init_kernel<<<(Bh * DECL * CINc + 255) / 256, 256, 0, stream>>>(
        seasF, trendF, meanE, seasI, trendI);

    // ---- encoder ----
    embed_kernel<<<Bh * SEQL, 256, 0, stream>>>(x_enc, x_mark_enc, enc_conv_w, enc_time_w, X, SEQL, CINc);
    for (int l = 0; l < 2; ++l) {
        const float* wl = enc_attn_w + (size_t)l * 4 * DD * DD;
        const float* bl = enc_attn_b + (size_t)l * 4 * DD;
        autocorr(X, X, wl, bl, Vb);           // attn out in Vb (free after agg)
        addv(X, Vb, Kb, S);
        decompL(Kb, X, TRJ, SEQL, DD);
        ffn(X, enc_ff1_w + (size_t)l * DD * DFFc, enc_ff2_w + (size_t)l * DFFc * DD, TMP);
        addv(X, TMP, Kb, S);
        decompL(Kb, X, TRJ, SEQL, DD);
    }
    layernorm_kernel<<<Bh * SEQL, 256, 0, stream>>>(X, enc_norm_g, enc_norm_b, TMP);
    sub_colmean_kernel<<<(Bh * DD + 255) / 256, 256, 0, stream>>>(TMP, ENC, SEQL);

    // ---- decoder ----
    embed_kernel<<<Bh * DECL, 256, 0, stream>>>(seasI, x_mark_dec, dec_conv_w, dec_time_w, X, DECL, CINc);
    autocorr(X, X, dec_self_w, dec_self_b, Vb);
    addv(X, Vb, Kb, S);
    decompL(Kb, X, T1, DECL, DD);
    autocorr(X, ENC, dec_cross_w, dec_cross_b, Vb);
    addv(X, Vb, Kb, S);
    decompL(Kb, X, T2, DECL, DD);
    ffn(X, dec_ff1_w, dec_ff2_w, TMP);
    addv(X, TMP, Kb, S);
    decompL(Kb, X, T3, DECL, DD);

    add3_kernel<<<(int)((S + 255) / 256), 256, 0, stream>>>(T1, T2, T3, TRJ, (int)S);
    trend_conv_kernel<<<Bh * DECL, 32, 0, stream>>>(TRJ, dec_trend_w, RT, DECL);
    addv(trendI, RT, RT, (size_t)Bh * DECL * CINc);

    layernorm_kernel<<<Bh * DECL, 256, 0, stream>>>(X, dec_norm_g, dec_norm_b, TMP);
    sub_colmean_kernel<<<(Bh * DD + 255) / 256, 256, 0, stream>>>(TMP, Q, DECL);

    final_kernel<<<Bh * PRD, 32, 0, stream>>>(Q, RT, proj_w, proj_b, out);
}
